// DGCNN_Cls_Encoder_48945447305572
// MI455X (gfx1250) — compile-verified, matrix operand order checked by reading, not run on hardware
//
#include <hip/hip_runtime.h>

// ---------------------------------------------------------------------------
// DGCNN encoder forward for MI455X (gfx1250, wave32, WMMA).
// All dense GEMMs use v_wmma_f32_16x16x32_f16 (f16 in, f32 accumulate).
// knn / BN stats / reductions stay fp32 scalar for ranking robustness.
// Working set (~170MB ws) is L2-resident (192MB, 23.3TB/s HBM irrelevant):
// compute/issue-bound -> f16 WMMA + b128 fragment loads + neighbor prefetch.
// ---------------------------------------------------------------------------

typedef __attribute__((ext_vector_type(16))) _Float16 v16h;
typedef __attribute__((ext_vector_type(8)))  float    v8f;
typedef __attribute__((ext_vector_type(4)))  float    v4f;

#define WMMA_F16(a, b, c) \
  __builtin_amdgcn_wmma_f32_16x16x32_f16(false, (a), false, (b), (short)0, (c), false, false)

// 16-bit A-matrix 16x32 fragment layout (ISA 7.12.2):
//   lane 0-15 : row M=lane,    K = {0..7, 16..23}
//   lane 16-31: row M=lane-16, K = {8..15, 24..31}
// half j<8 -> K = kb + j ; half j>=8 -> K = kb + 8 + j, kb = (lane&16)?8:0.
// Each lane therefore reads two contiguous 8-float runs -> 4x b128 loads.

__device__ __forceinline__ void cvt8(v16h& a, int jbase, const float* __restrict__ p) {
  const v4f u0 = *(const v4f*)p;
  const v4f u1 = *(const v4f*)(p + 4);
#pragma unroll
  for (int t = 0; t < 4; ++t) {
    a[jbase + t]     = (_Float16)u0[t];
    a[jbase + 4 + t] = (_Float16)u1[t];
  }
}

__device__ __forceinline__ void cvt8d(v16h& a, int jbase,
                                      const float* __restrict__ p0,
                                      const float* __restrict__ p1) {
  const v4f d0 = *(const v4f*)(p1) - *(const v4f*)(p0);
  const v4f d1 = *(const v4f*)(p1 + 4) - *(const v4f*)(p0 + 4);
#pragma unroll
  for (int t = 0; t < 4; ++t) {
    a[jbase + t]     = (_Float16)d0[t];
    a[jbase + 4 + t] = (_Float16)d1[t];
  }
}

// Vector path: requires 16B-aligned rows and CIN % 32 == 0.
__device__ __forceinline__ v16h load_a_frag_vec(const float* __restrict__ row, int kbase) {
  const int kb = (threadIdx.x & 16) ? 8 : 0;
  v16h a;
  cvt8(a, 0, row + kbase + kb);
  cvt8(a, 8, row + kbase + kb + 16);
  return a;
}

// Scalar masked path (layer 1: CIN=3, ld=3 / weight rows of 6 floats).
__device__ __forceinline__ v16h load_a_frag_s(const float* __restrict__ row,
                                              int kbase, int kvalid) {
  const int kb = (threadIdx.x & 16) ? 8 : 0;
  v16h a;
#pragma unroll
  for (int j = 0; j < 16; ++j) {
    int c = kbase + kb + ((j < 8) ? j : (j + 8));
    float v = (c < kvalid) ? row[c] : 0.0f;
    a[j] = (_Float16)v;
  }
  return a;
}

template <bool DIFF>
__device__ __forceinline__ v16h load_w_frag_vec(const float* __restrict__ w0,
                                                const float* __restrict__ w1,
                                                int kbase) {
  const int kb = (threadIdx.x & 16) ? 8 : 0;
  v16h b;
  if (DIFF) {
    cvt8d(b, 0, w0 + kbase + kb, w1 + kbase + kb);
    cvt8d(b, 8, w0 + kbase + kb + 16, w1 + kbase + kb + 16);
  } else {
    cvt8(b, 0, w0 + kbase + kb);
    cvt8(b, 8, w0 + kbase + kb + 16);
  }
  return b;
}

template <bool DIFF>
__device__ __forceinline__ v16h load_w_frag_s(const float* __restrict__ w0,
                                              const float* __restrict__ w1,
                                              int kbase, int kvalid) {
  const int kb = (threadIdx.x & 16) ? 8 : 0;
  v16h b;
#pragma unroll
  for (int j = 0; j < 16; ++j) {
    int c = kbase + kb + ((j < 8) ? j : (j + 8));
    float v = 0.0f;
    if (c < kvalid) v = DIFF ? (w1[c] - w0[c]) : w0[c];
    b[j] = (_Float16)v;
  }
  return b;
}

// ---------------------------------------------------------------------------
// knn: one thread per point, LDS-tiled candidates, fp32 distances,
// 20-deep stable insertion (ties -> lowest index, matching jax.lax.top_k).
// ---------------------------------------------------------------------------
template <int C>
__global__ __launch_bounds__(128) void knn_kernel(const float* __restrict__ x,
                                                  int ldx,
                                                  int* __restrict__ out_idx) {
  constexpr int KNN = 20, TM = 64, NPTS = 2048;
  __shared__ float tile[TM * C];
  const int tid = threadIdx.x;
  const int gpt = blockIdx.x * 128 + tid;      // global point id in [0, B*N)
  const int b = gpt >> 11;
  const int n = gpt & 2047;
  const float* xb = x + (size_t)b * NPTS * ldx;

  float xr[C];
#pragma unroll
  for (int c = 0; c < C; ++c) xr[c] = xb[(size_t)n * ldx + c];

  float topd[KNN];
  int topi[KNN];
#pragma unroll
  for (int j = 0; j < KNN; ++j) { topd[j] = -3.4e38f; topi[j] = 0; }

  for (int t = 0; t < NPTS; t += TM) {
    __syncthreads();
    for (int i = tid; i < TM * C; i += 128) {
      int mm = i / C, cc = i - mm * C;
      tile[i] = xb[(size_t)(t + mm) * ldx + cc];
    }
    __syncthreads();
    for (int mm = 0; mm < TM; ++mm) {
      const float* xm = &tile[mm * C];
      float s = 0.0f;
#pragma unroll
      for (int c = 0; c < C; ++c) { float d = xr[c] - xm[c]; s += d * d; }
      float dneg = -s;
      if (dneg > topd[KNN - 1]) {
        float dd = dneg; int ii = t + mm;
#pragma unroll
        for (int j = 0; j < KNN; ++j) {
          if (dd > topd[j]) {
            float td = topd[j]; int ti = topi[j];
            topd[j] = dd; topi[j] = ii; dd = td; ii = ti;
          }
        }
      }
    }
  }
#pragma unroll
  for (int j = 0; j < KNN; ++j) out_idx[(size_t)gpt * KNN + j] = topi[j];
}

// ---------------------------------------------------------------------------
// EdgeConv main pass: one wave per 16x16 output tile.
//   y[k] = neigh_k * Wa^T + ctr * (Wb - Wa)^T   (center term hoisted)
// Records ymax/ymin over k (monotone BN+lrelu+max trick) and per-tile column
// sums / sumsq (deterministic BN statistics). Next neighbor row prefetched.
// ---------------------------------------------------------------------------
template <int CIN, int COUT>
__global__ __launch_bounds__(32) void edge_pass_kernel(
    const float* __restrict__ xin, int ldin, const int* __restrict__ idx,
    const float* __restrict__ W, float* __restrict__ ymax,
    float* __restrict__ ymin, float* __restrict__ colsum,
    float* __restrict__ colsumsq) {
  constexpr bool VEC = (CIN % 32) == 0;  // layer 1 (CIN=3) -> scalar masked
  const int lane = threadIdx.x;
  const int tile = blockIdx.x;             // row tile over B*N (16 rows)
  const int ot = blockIdx.y;               // 16-wide output column tile
  const int mrow = lane & 15;              // A-role row within tile
  const int grow = tile * 16 + mrow;       // global row = b*N + n
  const int bbase = (grow >> 11) << 11;    // b * 2048
  const int ocol = ot * 16 + (lane & 15);  // B/D-role column
  const float* wrowA = W + (size_t)ocol * (2 * CIN);
  const float* wrowB = wrowA + CIN;
  constexpr int KC = (CIN + 31) / 32;

  // Hoist Wa fragments (reused for all 20 neighbors).
  v16h wa[KC];
#pragma unroll
  for (int t = 0; t < KC; ++t) {
    if constexpr (VEC) wa[t] = load_w_frag_vec<false>(wrowA, wrowA, t * 32);
    else               wa[t] = load_w_frag_s<false>(wrowA, wrowA, t * 32, CIN);
  }

  // s = ctr * (Wb - Wa)^T
  const float* ctr = xin + (size_t)grow * ldin;
  v8f s = {};
#pragma unroll
  for (int t = 0; t < KC; ++t) {
    v16h a, wd;
    if constexpr (VEC) {
      a = load_a_frag_vec(ctr, t * 32);
      wd = load_w_frag_vec<true>(wrowA, wrowB, t * 32);
    } else {
      a = load_a_frag_s(ctr, t * 32, CIN);
      wd = load_w_frag_s<true>(wrowA, wrowB, t * 32, CIN);
    }
    s = WMMA_F16(a, wd, s);
  }

  v8f ymx, ymn, sm, sq;
#pragma unroll
  for (int r = 0; r < 8; ++r) { ymx[r] = -3.4e38f; ymn[r] = 3.4e38f; sm[r] = 0.f; sq[r] = 0.f; }

  int j = idx[(size_t)grow * 20];  // first neighbor
  for (int k = 0; k < 20; ++k) {
    const float* nb = xin + (size_t)(bbase + j) * ldin;  // gathered neighbor row
    if (k < 19) {
      j = idx[(size_t)grow * 20 + k + 1];
      __builtin_prefetch(xin + (size_t)(bbase + j) * ldin, 0, 0);  // global_prefetch_b8
    }
    v8f acc = s;
#pragma unroll
    for (int t = 0; t < KC; ++t) {
      v16h a;
      if constexpr (VEC) a = load_a_frag_vec(nb, t * 32);
      else               a = load_a_frag_s(nb, t * 32, CIN);
      acc = WMMA_F16(a, wa[t], acc);
    }
#pragma unroll
    for (int r = 0; r < 8; ++r) {
      float v = acc[r];
      ymx[r] = fmaxf(ymx[r], v); ymn[r] = fminf(ymn[r], v);
      sm[r] += v; sq[r] += v * v;
    }
  }

  // D layout: lane<16 -> rows 0..7, lane>=16 -> rows 8..15, col = lane&15.
  const int rbase = (lane & 16) ? 8 : 0;
#pragma unroll
  for (int r = 0; r < 8; ++r) {
    size_t row = (size_t)tile * 16 + rbase + r;
    ymax[row * COUT + ocol] = ymx[r];
    ymin[row * COUT + ocol] = ymn[r];
  }
  float cs = 0.f, cq = 0.f;
#pragma unroll
  for (int r = 0; r < 8; ++r) { cs += sm[r]; cq += sq[r]; }
  cs += __shfl_xor(cs, 16, 32);
  cq += __shfl_xor(cq, 16, 32);
  if (lane < 16) {
    colsum[(size_t)tile * COUT + ocol] = cs;
    colsumsq[(size_t)tile * COUT + ocol] = cq;
  }
}

// ---------------------------------------------------------------------------
// Dense layer pass: y = A * W^T, one wave per 16x16 tile, raw y + tile stats.
// MODE 0: A[row*lda + c].   MODE 1 (L6): c<1024 -> glob[b][c], else cat.
// The 1024 split is 8-aligned, so every 8-float run stays on one source.
// ---------------------------------------------------------------------------
template <int COUT, int MODE>
__global__ __launch_bounds__(32) void linear_pass_kernel(
    const float* __restrict__ A, int lda, const float* __restrict__ G,
    const float* __restrict__ W, int KT, float* __restrict__ yraw,
    float* __restrict__ colsum, float* __restrict__ colsumsq) {
  const int lane = threadIdx.x;
  const int tile = blockIdx.x;
  const int ot = blockIdx.y;
  const int grow = tile * 16 + (lane & 15);
  const int ocol = ot * 16 + (lane & 15);
  const float* wrow = W + (size_t)ocol * KT;
  const int kb2 = (lane & 16) ? 8 : 0;
  const float* arow = A + (size_t)grow * lda;
  const float* grb = (MODE == 1) ? (G + (size_t)(grow >> 11) * 1024) : nullptr;

  v8f acc = {};
  for (int kb = 0; kb < KT; kb += 32) {
    v16h a;
    if (MODE == 1) {
      const int c0 = kb + kb2;        // 8-aligned group bases: c0, c0+16
      const float* p0 = (c0 < 1024) ? (grb + c0) : (arow + (c0 - 1024));
      const int c1 = c0 + 16;
      const float* p1 = (c1 < 1024) ? (grb + c1) : (arow + (c1 - 1024));
      cvt8(a, 0, p0);
      cvt8(a, 8, p1);
    } else {
      a = load_a_frag_vec(arow, kb);
    }
    v16h b = load_w_frag_vec<false>(wrow, wrow, kb);
    acc = WMMA_F16(a, b, acc);
  }

  const int rbase = (lane & 16) ? 8 : 0;
  float cs = 0.f, cq = 0.f;
#pragma unroll
  for (int r = 0; r < 8; ++r) {
    float v = acc[r];
    yraw[((size_t)tile * 16 + rbase + r) * COUT + ocol] = v;
    cs += v; cq += v * v;
  }
  cs += __shfl_xor(cs, 16, 32);
  cq += __shfl_xor(cq, 16, 32);
  if (lane < 16) {
    colsum[(size_t)tile * COUT + ocol] = cs;
    colsumsq[(size_t)tile * COUT + ocol] = cq;
  }
}

// ---------------------------------------------------------------------------
// BN helpers (deterministic reduction over fixed tile order), apply kernels.
// ---------------------------------------------------------------------------
__global__ void bn_finalize_kernel(const float* __restrict__ colsum,
                                   const float* __restrict__ colsumsq,
                                   const float* __restrict__ g,
                                   const float* __restrict__ bias,
                                   float* __restrict__ bn_a,
                                   float* __restrict__ bn_c, int C, int T,
                                   float invcnt) {
  int o = blockIdx.x * blockDim.x + threadIdx.x;
  if (o >= C) return;
  float s = 0.f, q = 0.f;
  for (int t = 0; t < T; ++t) {
    s += colsum[(size_t)t * C + o];
    q += colsumsq[(size_t)t * C + o];
  }
  float mu = s * invcnt;
  float var = q * invcnt - mu * mu;
  float a = g[o] * rsqrtf(var + 1e-5f);
  bn_a[o] = a;
  bn_c[o] = bias[o] - mu * a;
}

// max_k lrelu(a*y+b) == lrelu(a*(a>=0 ? ymax : ymin) + b)  (monotone in y).
__global__ void edge_apply_kernel(const float* __restrict__ ymax,
                                  const float* __restrict__ ymin,
                                  const float* __restrict__ bn_a,
                                  const float* __restrict__ bn_c,
                                  float* __restrict__ cat, int colOff, int C,
                                  int total) {
  int i = blockIdx.x * 256 + threadIdx.x;
  if (i >= total) return;
  int row = i / C, o = i - row * C;
  float a = bn_a[o];
  float v = (a >= 0.f) ? ymax[i] : ymin[i];
  float y = a * v + bn_c[o];
  cat[(size_t)row * 512 + colOff + o] = (y >= 0.f) ? y : 0.2f * y;
}

__global__ void bn_lrelu_kernel(float* __restrict__ y,
                                const float* __restrict__ bn_a,
                                const float* __restrict__ bn_c, int C,
                                int total) {
  int i = blockIdx.x * 256 + threadIdx.x;
  if (i >= total) return;
  int o = i % C;
  float v = bn_a[o] * y[i] + bn_c[o];
  y[i] = (v >= 0.f) ? v : 0.2f * v;
}

__global__ void rowmax_kernel(const float* __restrict__ h,
                              float* __restrict__ glob) {
  // h: (8,2048,1024) -> glob: (8,1024). 32 blocks x 256 threads, coalesced.
  int b = blockIdx.x >> 2;
  int o = ((blockIdx.x & 3) << 8) + threadIdx.x;
  const float* p = h + (size_t)b * 2048 * 1024 + o;
  float m = -3.4e38f;
  for (int n = 0; n < 2048; ++n) m = fmaxf(m, p[(size_t)n * 1024]);
  glob[b * 1024 + o] = m;
}

// ---------------------------------------------------------------------------
extern "C" void kernel_launch(void* const* d_in, const int* in_sizes, int n_in,
                              void* d_out, int out_size, void* d_ws,
                              size_t ws_size, hipStream_t stream) {
  (void)in_sizes; (void)n_in; (void)out_size; (void)ws_size;
  const float* x = (const float*)d_in[0];
  const float *W1 = (const float*)d_in[1],  *g1 = (const float*)d_in[2],  *b1 = (const float*)d_in[3];
  const float *W2 = (const float*)d_in[4],  *g2 = (const float*)d_in[5],  *b2 = (const float*)d_in[6];
  const float *W3 = (const float*)d_in[7],  *g3 = (const float*)d_in[8],  *b3 = (const float*)d_in[9];
  const float *W4 = (const float*)d_in[10], *g4 = (const float*)d_in[11], *b4 = (const float*)d_in[12];
  const float *W5 = (const float*)d_in[13], *g5 = (const float*)d_in[14], *b5 = (const float*)d_in[15];
  const float *W6 = (const float*)d_in[16], *g6 = (const float*)d_in[17], *b6 = (const float*)d_in[18];
  const float *W7 = (const float*)d_in[19], *g7 = (const float*)d_in[20], *b7 = (const float*)d_in[21];
  float* out = (float*)d_out;

  const int BN = 8 * 2048;  // 16384 rows
  const int T = BN / 16;    // 1024 row tiles

  // Workspace layout (~170 MB, all write-before-read; harness poison-safe).
  char* ws = (char*)d_ws;
  size_t off = 0;
  auto alloc = [&](size_t bytes) {
    void* p = ws + off;
    off += (bytes + 255) & ~(size_t)255;
    return p;
  };
  int*   idx      = (int*)alloc((size_t)BN * 20 * 4);
  float* cat      = (float*)alloc((size_t)BN * 512 * 4);   // [x1|x2|x3|x4]
  float* ymax     = (float*)alloc((size_t)BN * 256 * 4);
  float* ymin     = (float*)alloc((size_t)BN * 256 * 4);
  float* colsum   = (float*)alloc((size_t)T * 1024 * 4);
  float* colsumsq = (float*)alloc((size_t)T * 1024 * 4);
  float* bn_a     = (float*)alloc(1024 * 4);
  float* bn_c     = (float*)alloc(1024 * 4);
  float* ybuf     = (float*)alloc((size_t)BN * 1024 * 4);  // y5 -> h (in place)
  float* glob     = (float*)alloc(8 * 1024 * 4);
  float* h2       = (float*)alloc((size_t)BN * 512 * 4);   // y6 -> h2 (in place)

  const float inv_k = 1.0f / (float)(BN * 20);
  const float inv_n = 1.0f / (float)BN;

  // ---- EdgeConv 1: x(3) -> cat[:, 0:64]
  knn_kernel<3><<<BN / 128, 128, 0, stream>>>(x, 3, idx);
  edge_pass_kernel<3, 64><<<dim3(T, 4), 32, 0, stream>>>(x, 3, idx, W1, ymax, ymin, colsum, colsumsq);
  bn_finalize_kernel<<<1, 256, 0, stream>>>(colsum, colsumsq, g1, b1, bn_a, bn_c, 64, T, inv_k);
  edge_apply_kernel<<<(BN * 64) / 256, 256, 0, stream>>>(ymax, ymin, bn_a, bn_c, cat, 0, 64, BN * 64);

  // ---- EdgeConv 2: cat[:,0:64] -> cat[:, 64:128]
  knn_kernel<64><<<BN / 128, 128, 0, stream>>>(cat, 512, idx);
  edge_pass_kernel<64, 64><<<dim3(T, 4), 32, 0, stream>>>(cat, 512, idx, W2, ymax, ymin, colsum, colsumsq);
  bn_finalize_kernel<<<1, 256, 0, stream>>>(colsum, colsumsq, g2, b2, bn_a, bn_c, 64, T, inv_k);
  edge_apply_kernel<<<(BN * 64) / 256, 256, 0, stream>>>(ymax, ymin, bn_a, bn_c, cat, 64, 64, BN * 64);

  // ---- EdgeConv 3: cat[:,64:128] -> cat[:, 128:256]
  knn_kernel<64><<<BN / 128, 128, 0, stream>>>(cat + 64, 512, idx);
  edge_pass_kernel<64, 128><<<dim3(T, 8), 32, 0, stream>>>(cat + 64, 512, idx, W3, ymax, ymin, colsum, colsumsq);
  bn_finalize_kernel<<<1, 256, 0, stream>>>(colsum, colsumsq, g3, b3, bn_a, bn_c, 128, T, inv_k);
  edge_apply_kernel<<<(BN * 128) / 256, 256, 0, stream>>>(ymax, ymin, bn_a, bn_c, cat, 128, 128, BN * 128);

  // ---- EdgeConv 4: cat[:,128:256] -> cat[:, 256:512]
  knn_kernel<128><<<BN / 128, 128, 0, stream>>>(cat + 128, 512, idx);
  edge_pass_kernel<128, 256><<<dim3(T, 16), 32, 0, stream>>>(cat + 128, 512, idx, W4, ymax, ymin, colsum, colsumsq);
  bn_finalize_kernel<<<1, 256, 0, stream>>>(colsum, colsumsq, g4, b4, bn_a, bn_c, 256, T, inv_k);
  edge_apply_kernel<<<(BN * 256) / 256, 256, 0, stream>>>(ymax, ymin, bn_a, bn_c, cat, 256, 256, BN * 256);

  // ---- L5: cat(512) -> h(1024), BN+lrelu, global max over n
  linear_pass_kernel<1024, 0><<<dim3(T, 64), 32, 0, stream>>>(cat, 512, nullptr, W5, 512, ybuf, colsum, colsumsq);
  bn_finalize_kernel<<<4, 256, 0, stream>>>(colsum, colsumsq, g5, b5, bn_a, bn_c, 1024, T, inv_n);
  bn_lrelu_kernel<<<(BN * 1024) / 256, 256, 0, stream>>>(ybuf, bn_a, bn_c, 1024, BN * 1024);
  rowmax_kernel<<<32, 256, 0, stream>>>(ybuf, glob);

  // ---- L6: [glob(1024) | cat(512)] -> h2(512), BN+lrelu
  linear_pass_kernel<512, 1><<<dim3(T, 32), 32, 0, stream>>>(cat, 512, glob, W6, 1536, h2, colsum, colsumsq);
  bn_finalize_kernel<<<2, 256, 0, stream>>>(colsum, colsumsq, g6, b6, bn_a, bn_c, 512, T, inv_n);
  bn_lrelu_kernel<<<(BN * 512) / 256, 256, 0, stream>>>(h2, bn_a, bn_c, 512, BN * 512);

  // ---- L7: h2(512) -> out(128), BN+lrelu (in place on d_out)
  linear_pass_kernel<128, 0><<<dim3(T, 8), 32, 0, stream>>>(h2, 512, nullptr, W7, 512, out, colsum, colsumsq);
  bn_finalize_kernel<<<1, 256, 0, stream>>>(colsum, colsumsq, g7, b7, bn_a, bn_c, 128, T, inv_n);
  bn_lrelu_kernel<<<(BN * 128) / 256, 256, 0, stream>>>(out, bn_a, bn_c, 128, BN * 128);
}
